// UNet_13365938225631
// MI455X (gfx1250) — matrix operand order repaired and checked by reading, not compiled, not run on hardware
//
#include <hip/hip_runtime.h>

// MI455X / gfx1250 sparse U-Net.
// Heavy gather-GEMMs run on v_wmma_f32_16x16x32_f16 (8192 MACs/instr, f32
// accumulate). All live data fits the 192MB L2, so the matrix-op ceiling, not
// the 23.3 TB/s HBM, is the limit -> f16 inputs + f32 accumulation, and
// maximize WMMA issued per load: 4 M-tiles per block share each B fragment.

typedef __attribute__((ext_vector_type(16))) _Float16 v16h;
typedef __attribute__((ext_vector_type(8)))  float    v8f;

union F16Frag {
    v16h v;
    uint4 q[2];
};

// ---------------- BN(inference) + ReLU + f32->f16, writes padded (N+1) rows,
// row N is the zero row used by sentinel gather indices.
// One block per row, blockDim == C: no div/mod, fully coalesced. ----------------
__global__ void bn_relu_f16_kernel(const float* __restrict__ x, int rowStride,
                                   const float* __restrict__ g, const float* __restrict__ b,
                                   const float* __restrict__ m, const float* __restrict__ vv,
                                   _Float16* __restrict__ out, int N)
{
    const int row = blockIdx.x;
    const int col = threadIdx.x;
    const int C   = blockDim.x;
    float val = 0.f;
    if (row < N) {
        float xv = x[(long long)row * rowStride + col];
        float r  = 1.f / sqrtf(vv[col] + 1e-4f);
        val = fmaxf((xv - m[col]) * r * g[col] + b[col], 0.f);
    }
    out[(long long)row * C + col] = (_Float16)val;
}

// ---------------- Weight prep: f32 [K][Cin][Cout] -> f16 [K][Cout][Cin] so a
// lane's 16-K B fragment is 32 contiguous bytes. Tiny, runs once per call. ----
__global__ void wprep_kernel(const float* __restrict__ w, _Float16* __restrict__ wt,
                             int K, int CIN, int COUT)
{
    long long i = (long long)blockIdx.x * blockDim.x + threadIdx.x;
    long long total = (long long)K * CIN * COUT;
    if (i >= total) return;
    int k  = (int)(i / ((long long)CIN * COUT));
    int r  = (int)(i % ((long long)CIN * COUT));
    int ci = r / COUT;
    int co = r % COUT;
    wt[((long long)k * COUT + co) * CIN + ci] = (_Float16)w[i];
}

// ---------------- Sparse conv: out[m,:] = sum_k pad(act)[idx[m,k],:] @ W[k]
// One block = MT*16 output rows; wave w owns output columns [16w,16w+16) and
// keeps MT v8f accumulators. Per tap: stage MT*16 gathered rows in LDS (b128),
// then per 32-K chunk load one B fragment and issue MT WMMAs against it. ------
template<int CIN, int COUT, int K, int MT>
__global__ __launch_bounds__(2 * COUT)
void sconv_wmma_kernel(const _Float16* __restrict__ act,   // (Nsrc+1) x CIN, row Nsrc == 0
                       const int* __restrict__ idx,        // Nout x K (sentinel = Nsrc)
                       const _Float16* __restrict__ wt,    // K x COUT x CIN (transposed f16)
                       float* __restrict__ out, int Nout, int Nsrc,
                       int outStride, int outOffset)
{
    constexpr int NWAVES = COUT / 16;
    constexpr int NT     = 32 * NWAVES;
    constexpr int CHUNKS = CIN / 32;
    constexpr int VEC    = CIN / 8;        // 16-byte vectors per row
    constexpr int ROWS   = 16 * MT;

    __shared__ alignas(16) _Float16 As[ROWS * CIN];
    __shared__ int sIdx[ROWS];

    const int tid  = threadIdx.x;
    const int lane = tid & 31;
    const int wave = tid >> 5;
    const int m0   = blockIdx.x * ROWS;
    const int n0   = wave * 16;
    const int ncol = lane & 15;            // B/D column owned by this lane
    const int hi   = lane >> 4;            // lane half selects K sub-range

    v8f acc[MT] = {};                      // MT 16x16 f32 accumulators

    for (int k = 0; k < K; ++k) {
        __syncthreads();                   // protect As of previous tap
        for (int t = tid; t < ROWS; t += NT) {
            int row = m0 + t;
            sIdx[t] = (row < Nout) ? idx[(long long)row * K + k] : Nsrc;
        }
        __syncthreads();
        // Cooperative gather of ROWS x CIN f16 into LDS (b128 granules).
        for (int t = tid; t < ROWS * VEC; t += NT) {
            int r = t / VEC;
            int c = t % VEC;
            const uint4* src =
                reinterpret_cast<const uint4*>(act + (long long)sIdx[r] * CIN) + c;
            reinterpret_cast<uint4*>(As + r * CIN)[c] = *src;
        }
        __syncthreads();

        const _Float16* wrow = wt + ((long long)k * COUT + n0 + ncol) * CIN;
#pragma unroll
        for (int c = 0; c < CHUNKS; ++c) {
            // B 32x16 f16 layout: lane owns column N=lane&15;
            // lanes 0-15 hold K 0..15, lanes 16-31 hold K 16..31 (contiguous).
            F16Frag b;
            const _Float16* bp = wrow + c * 32 + hi * 16;
            b.q[0] = *reinterpret_cast<const uint4*>(bp);
            b.q[1] = *reinterpret_cast<const uint4*>(bp + 8);
#pragma unroll
            for (int mt = 0; mt < MT; ++mt) {
                // A 16x32 f16 layout: lane<16 -> M=lane, K 0..15;
                //                     lane>=16 -> M=lane-16, K 16..31 (contiguous).
                F16Frag a;
                const _Float16* ap = As + (mt * 16 + ncol) * CIN + c * 32 + hi * 16;
                a.q[0] = *reinterpret_cast<const uint4*>(ap);        // K 0..7 (+hi*16)
                a.q[1] = *reinterpret_cast<const uint4*>(ap + 8);    // K 8..15 (+hi*16)
                acc[mt] = __builtin_amdgcn_wmma_f32_16x16x32_f16(false, a.v, false, b.v,
                                                                 (short)0, acc[mt],
                                                                 false, false);
            }
        }
    }

    // C/D layout: lane = column n, VGPR i = row i + 8*(lane>=16).
#pragma unroll
    for (int mt = 0; mt < MT; ++mt) {
        const int base = m0 + mt * 16 + hi * 8;
#pragma unroll
        for (int i = 0; i < 8; ++i) {
            int row = base + i;
            if (row < Nout)
                out[(long long)row * outStride + outOffset + (n0 + ncol)] = acc[mt][i];
        }
    }
}

// ---------------- Stride-2 transpose conv: out[m,n] = sum_c x[parent[m],c]*W[slot[m],c,n]
// ~2 GF total -> fp32 VALU. Stage R parent rows in LDS so each row is read
// from global once instead of COUT times. ----------------
template<int CIN, int COUT>
__global__ __launch_bounds__(256)
void deconv_kernel(const _Float16* __restrict__ act, const int* __restrict__ parent,
                   const int* __restrict__ slot, const float* __restrict__ W,
                   float* __restrict__ out, int N, int outStride, int outOffset)
{
    constexpr int R   = 256 / COUT;        // rows per block
    constexpr int VEC = CIN / 8;
    __shared__ alignas(16) _Float16 xs[R * CIN];
    __shared__ int sPar[R];
    __shared__ int sSlot[R];

    const int tid = threadIdx.x;
    const int m0  = blockIdx.x * R;
    if (tid < R) {
        int m = m0 + tid;
        sPar[tid]  = (m < N) ? parent[m] : 0;
        sSlot[tid] = (m < N) ? slot[m]   : 0;
    }
    __syncthreads();
    for (int t = tid; t < R * VEC; t += 256) {
        int r = t / VEC;
        int c = t % VEC;
        reinterpret_cast<uint4*>(xs + r * CIN)[c] =
            reinterpret_cast<const uint4*>(act + (long long)sPar[r] * CIN)[c];
    }
    __syncthreads();

    const int r = tid / COUT;
    const int n = tid % COUT;
    const int m = m0 + r;
    if (m >= N) return;
    const float*    wr = W + (long long)sSlot[r] * CIN * COUT + n;
    const _Float16* xr = xs + r * CIN;
    float acc = 0.f;
#pragma unroll 8
    for (int c = 0; c < CIN; ++c)
        acc = fmaf((float)xr[c], wr[c * COUT], acc);
    out[(long long)m * outStride + outOffset + n] = acc;
}

// =====================================================================
extern "C" void kernel_launch(void* const* d_in, const int* in_sizes, int n_in,
                              void* d_out, int out_size, void* d_ws, size_t ws_size,
                              hipStream_t stream)
{
    // ---- input unpack (setup_inputs dict order, nested lists flattened) ----
    const float* feat = (const float*)d_in[0];
    const float* bnp[9][4];
    for (int l = 0; l < 9; ++l)
        for (int j = 0; j < 4; ++j)
            bnp[l][j] = (const float*)d_in[1 + l * 4 + j];
    const float* sub_w[5];  for (int i = 0; i < 5; ++i) sub_w[i]  = (const float*)d_in[37 + i];
    const float* down_w[2]; for (int i = 0; i < 2; ++i) down_w[i] = (const float*)d_in[42 + i];
    const float* up_w[2];   for (int i = 0; i < 2; ++i) up_w[i]   = (const float*)d_in[44 + i];
    const int* sub_idx0   = (const int*)d_in[46];
    const int* sub_idx1   = (const int*)d_in[47];
    const int* sub_idx2   = (const int*)d_in[48];
    const int* down_idx0  = (const int*)d_in[49];
    const int* down_idx1  = (const int*)d_in[50];
    const int* up_parent0 = (const int*)d_in[51];
    const int* up_parent1 = (const int*)d_in[52];
    const int* up_slot0   = (const int*)d_in[53];
    const int* up_slot1   = (const int*)d_in[54];

    const int N0 = in_sizes[51];
    const int N1 = in_sizes[52];
    const int N2 = in_sizes[48] / 27;

    // ---- workspace bump allocator (lifetime-aliased) ----
    char* ws = (char*)d_ws;
    size_t off = 0;
    auto alloc = [&](size_t bytes) -> void* {
        off = (off + 255) & ~(size_t)255;
        void* p = ws + off;
        off += bytes;
        return p;
    };

    // f16 transposed weights (persistent, ~2MB)
    _Float16* wt0  = (_Float16*)alloc((size_t)27 * 32  * 32  * 2);
    _Float16* wt1  = (_Float16*)alloc((size_t)27 * 64  * 64  * 2);
    _Float16* wt2  = (_Float16*)alloc((size_t)27 * 128 * 128 * 2);
    _Float16* wt3  = (_Float16*)alloc((size_t)27 * 128 * 64  * 2);
    _Float16* wt4  = (_Float16*)alloc((size_t)27 * 64  * 32  * 2);
    _Float16* wtd0 = (_Float16*)alloc((size_t)8  * 32  * 64  * 2);
    _Float16* wtd1 = (_Float16*)alloc((size_t)8  * 64  * 128 * 2);

    // persistent concat buffers (skip connections live in-place)
    float* concat0 = (float*)alloc((size_t)N0 * 64  * 4);  // [skip0 | up1]
    float* concat1 = (float*)alloc((size_t)N1 * 128 * 4);  // [skip1 | up0]

    // shared f32 temp region: f_down0 / {f_down1,f_sub2} / f_sub3 (disjoint lifetimes)
    size_t tmpFloats = (size_t)N1 * 64;
    if ((size_t)2 * N2 * 128 > tmpFloats) tmpFloats = (size_t)2 * N2 * 128;
    float* tmp = (float*)alloc(tmpFloats * 4);
    float* f_down0 = tmp;
    float* f_down1 = tmp;
    float* f_sub2  = tmp + (size_t)N2 * 128;
    float* f_sub3  = tmp;

    // shared f16 activation region (exactly one live at any time)
    size_t actHalfs = (size_t)(N0 + 1) * 64;
    if ((size_t)(N1 + 1) * 128 > actHalfs) actHalfs = (size_t)(N1 + 1) * 128;
    if ((size_t)(N2 + 1) * 128 > actHalfs) actHalfs = (size_t)(N2 + 1) * 128;
    _Float16* act = (_Float16*)alloc(actHalfs * 2);

    auto blks = [](long long n, int bs) { return (unsigned)((n + bs - 1) / bs); };

    // ---- weight prep ----
    {
        struct { const float* w; _Float16* wt; int K, CIN, COUT; } jobs[7] = {
            { sub_w[0],  wt0,  27, 32,  32  }, { sub_w[1],  wt1,  27, 64,  64  },
            { sub_w[2],  wt2,  27, 128, 128 }, { sub_w[3],  wt3,  27, 128, 64  },
            { sub_w[4],  wt4,  27, 64,  32  }, { down_w[0], wtd0, 8,  32,  64  },
            { down_w[1], wtd1, 8,  64,  128 },
        };
        for (auto& j : jobs) {
            long long tot = (long long)j.K * j.CIN * j.COUT;
            wprep_kernel<<<blks(tot, 256), 256, 0, stream>>>(j.w, j.wt, j.K, j.CIN, j.COUT);
        }
    }

    auto bn = [&](const float* x, int stride, int l, int N, int C) {
        bn_relu_f16_kernel<<<(unsigned)(N + 1), C, 0, stream>>>(
            x, stride, bnp[l][0], bnp[l][1], bnp[l][2], bnp[l][3], act, N);
    };

    // ---- level 0 down ----
    bn(feat, 32, 0, N0, 32);
    sconv_wmma_kernel<32, 32, 27, 4><<<blks(N0, 64), 64, 0, stream>>>(
        act, sub_idx0, wt0, concat0, N0, N0, 64, 0);             // skip0 -> concat0[:, :32]
    bn(concat0, 64, 1, N0, 32);
    sconv_wmma_kernel<32, 64, 8, 4><<<blks(N1, 64), 128, 0, stream>>>(
        act, down_idx0, wtd0, f_down0, N1, N0, 64, 0);
    // ---- level 1 down ----
    bn(f_down0, 64, 2, N1, 64);
    sconv_wmma_kernel<64, 64, 27, 4><<<blks(N1, 64), 128, 0, stream>>>(
        act, sub_idx1, wt1, concat1, N1, N1, 128, 0);            // skip1 -> concat1[:, :64]
    bn(concat1, 128, 3, N1, 64);
    sconv_wmma_kernel<64, 128, 8, 4><<<blks(N2, 64), 256, 0, stream>>>(
        act, down_idx1, wtd1, f_down1, N2, N1, 128, 0);
    // ---- level 2 bottleneck ----
    bn(f_down1, 128, 4, N2, 128);
    sconv_wmma_kernel<128, 128, 27, 4><<<blks(N2, 64), 256, 0, stream>>>(
        act, sub_idx2, wt2, f_sub2, N2, N2, 128, 0);
    // ---- up to level 1 ----
    bn(f_sub2, 128, 5, N2, 128);
    deconv_kernel<128, 64><<<blks(N1, 4), 256, 0, stream>>>(
        act, up_parent1, up_slot1, up_w[0], concat1, N1, 128, 64);
    bn(concat1, 128, 6, N1, 128);
    sconv_wmma_kernel<128, 64, 27, 4><<<blks(N1, 64), 128, 0, stream>>>(
        act, sub_idx1, wt3, f_sub3, N1, N1, 64, 0);
    // ---- up to level 0 ----
    bn(f_sub3, 64, 7, N1, 64);
    deconv_kernel<64, 32><<<blks(N0, 8), 256, 0, stream>>>(
        act, up_parent0, up_slot0, up_w[1], concat0, N0, 64, 32);
    bn(concat0, 64, 8, N0, 64);
    sconv_wmma_kernel<64, 32, 27, 4><<<blks(N0, 64), 64, 0, stream>>>(
        act, sub_idx0, wt4, (float*)d_out, N0, N0, 32, 0);

    (void)n_in; (void)out_size; (void)ws_size;
}
